// Linear_41334765257142
// MI455X (gfx1250) — compile-verified, matrix operand order checked by reading, not run on hardware
//
#include <hip/hip_runtime.h>

// ---- types for WMMA ----
typedef __attribute__((ext_vector_type(16))) __bf16 v16bf;
typedef __attribute__((ext_vector_type(8)))  __bf16 v8bf;
typedef __attribute__((ext_vector_type(4)))  __bf16 v4bf;
typedef __attribute__((ext_vector_type(8)))  float  v8f;

// Problem dims fixed by the reference harness.
constexpr int T_DIM = 8192;    // tokens (M)
constexpr int I_DIM = 4096;    // in_features (K)
constexpr int O_DIM = 14336;   // out_features (N)
constexpr int QBLK  = 128;     // dequant block size

constexpr int BM = 128;        // block tile M
constexpr int BN = 128;        // block tile N
constexpr int BK = 32;         // k tile == one wmma_..x32 step
constexpr int LDSS = BK + 8;   // padded bf16 row stride (80B: 16B-aligned, bank-spread)

union Frag16 { v16bf v; v8bf h[2]; };

// ---- global -> registers: one 128x32 f32 tile each for x and w ----
__device__ __forceinline__ void load_tiles_g2r(const float* __restrict__ x,
                                               const float* __restrict__ w,
                                               int t0, int n0, int k0,
                                               int lrow, int cg,
                                               float4 ra[4], float4 rb[4]) {
#pragma unroll
    for (int rr = 0; rr < 4; ++rr) {
        const int r = lrow + rr * 32;
        ra[rr] = *reinterpret_cast<const float4*>(
            &x[(size_t)(t0 + r) * I_DIM + k0 + cg * 4]);
        rb[rr] = *reinterpret_cast<const float4*>(
            &w[(size_t)(n0 + r) * I_DIM + k0 + cg * 4]);
    }
}

// ---- registers -> LDS: convert to bf16, dequant-scale the weight tile ----
__device__ __forceinline__ void stage_r2lds(const float4 ra[4], const float4 rb[4],
                                            float s, __bf16* la, __bf16* lb,
                                            int lrow, int cg) {
#pragma unroll
    for (int rr = 0; rr < 4; ++rr) {
        const int r = lrow + rr * 32;
        v4bf pa, pb;
        pa[0] = (__bf16)ra[rr].x; pa[1] = (__bf16)ra[rr].y;
        pa[2] = (__bf16)ra[rr].z; pa[3] = (__bf16)ra[rr].w;
        pb[0] = (__bf16)(rb[rr].x * s); pb[1] = (__bf16)(rb[rr].y * s);
        pb[2] = (__bf16)(rb[rr].z * s); pb[3] = (__bf16)(rb[rr].w * s);
        *reinterpret_cast<v4bf*>(&la[r * LDSS + cg * 4]) = pa;
        *reinterpret_cast<v4bf*>(&lb[r * LDSS + cg * 4]) = pb;
    }
}

__global__ __launch_bounds__(256)
void fp8block_dequant_gemm(const float* __restrict__ x,
                           const float* __restrict__ w,
                           const float* __restrict__ scale,
                           float* __restrict__ out)
{
    // double-buffered tiles: 2 * (128*40*2B) * 2 operands = 40 KB of 320 KB WGP LDS
    __shared__ __bf16 lds_a[2][BM * LDSS];   // x tile   [row=m][k]
    __shared__ __bf16 lds_b[2][BN * LDSS];   // w tile   [row=n][k] (dequantized)

    const int tid   = threadIdx.x;
    const int lane  = tid & 31;
    const int wave  = tid >> 5;      // 0..7
    const int wm    = wave >> 2;     // 0..1 : wave row in 2x4 wave grid
    const int wn    = wave & 3;      // 0..3 : wave col

    const int t0 = blockIdx.y * BM;  // global M base
    const int n0 = blockIdx.x * BN;  // global N base

    // global->LDS loader mapping: 256 threads, float4 per access
    const int cg   = tid & 7;        // column group of 4 floats (8 * 4 = 32 = BK)
    const int lrow = tid >> 3;       // 0..31, stride 32 over 128 rows

    const int khalf = lane >> 4;     // 0/1 : which K-half this lane holds
    const int l16   = lane & 15;     // row (A) / col (B) within 16x16 tile

    const int srow = blockIdx.x * (I_DIM / QBLK);  // scale row base for this N tile

    v8f acc[4][2];
#pragma unroll
    for (int i = 0; i < 4; ++i)
#pragma unroll
        for (int j = 0; j < 2; ++j)
            acc[i][j] = (v8f){0.f, 0.f, 0.f, 0.f, 0.f, 0.f, 0.f, 0.f};

    // ---- prologue: stage k-tile 0 into buffer 0 ----
    {
        float4 ra[4], rb[4];
        load_tiles_g2r(x, w, t0, n0, 0, lrow, cg, ra, rb);
        stage_r2lds(ra, rb, scale[srow], lds_a[0], lds_b[0], lrow, cg);
    }
    __syncthreads();

    int cur = 0;
    for (int k0 = 0; k0 < I_DIM; k0 += BK) {
        const int  kn       = k0 + BK;
        const bool has_next = kn < I_DIM;

        // ---- issue next tile's global loads before touching LDS (latency overlap) ----
        float4 ra[4], rb[4];
        float  sn = 0.f;
        if (has_next) {
            sn = scale[srow + (kn >> 7)];
            load_tiles_g2r(x, w, t0, n0, kn, lrow, cg, ra, rb);
        }

        // ---- build fragments from LDS (ISA 7.12.2 16-bit A/B layout) ----
        const __bf16* la = lds_a[cur];
        const __bf16* lb = lds_b[cur];
        Frag16 af[4], bfm[2];
#pragma unroll
        for (int i = 0; i < 4; ++i) {
            const int row = wm * 64 + i * 16 + l16;
            const __bf16* base = &la[row * LDSS + khalf * 8];
            af[i].h[0] = *reinterpret_cast<const v8bf*>(base);       // K = h*8 .. +7
            af[i].h[1] = *reinterpret_cast<const v8bf*>(base + 16);  // K = h*8+16 ..
        }
#pragma unroll
        for (int j = 0; j < 2; ++j) {
            const int col = wn * 32 + j * 16 + l16;
            const __bf16* base = &lb[col * LDSS + khalf * 8];
            bfm[j].h[0] = *reinterpret_cast<const v8bf*>(base);
            bfm[j].h[1] = *reinterpret_cast<const v8bf*>(base + 16);
        }

        // ---- 8 WMMAs: 64x32 wave tile, f32 accumulate ----
#pragma unroll
        for (int i = 0; i < 4; ++i)
#pragma unroll
            for (int j = 0; j < 2; ++j)
                acc[i][j] = __builtin_amdgcn_wmma_f32_16x16x32_bf16(
                    false, af[i].v, false, bfm[j].v,
                    (short)0, acc[i][j], false, false);

        // ---- drain next tile into the other buffer; one barrier per k-tile ----
        if (has_next)
            stage_r2lds(ra, rb, sn, lds_a[cur ^ 1], lds_b[cur ^ 1], lrow, cg);
        __syncthreads();
        cur ^= 1;
    }

    // ---- epilogue: C/D layout = VGPR r -> M = r + 8*(lane>=16), N = lane&15 ----
#pragma unroll
    for (int i = 0; i < 4; ++i) {
        const int row = t0 + wm * 64 + i * 16 + khalf * 8;
#pragma unroll
        for (int j = 0; j < 2; ++j) {
            const int col = n0 + wn * 32 + j * 16 + l16;
#pragma unroll
            for (int r = 0; r < 8; ++r)
                out[(size_t)(row + r) * O_DIM + col] = acc[i][j][r];
        }
    }
}

extern "C" void kernel_launch(void* const* d_in, const int* in_sizes, int n_in,
                              void* d_out, int out_size, void* d_ws, size_t ws_size,
                              hipStream_t stream) {
    const float* x     = (const float*)d_in[0];
    const float* w     = (const float*)d_in[1];
    const float* scale = (const float*)d_in[2];
    float* out         = (float*)d_out;

    dim3 grid(O_DIM / BN, T_DIM / BM);   // 112 x 64 workgroups
    fp8block_dequant_gemm<<<grid, 256, 0, stream>>>(x, w, scale, out);
}
